// ChessNet_2216203125401
// MI455X (gfx1250) — matrix operand order repaired and checked by reading, not compiled, not run on hardware
//
// MI455X (gfx1250, wave32, WMMA) forward pass for the LC0-style chess transformer.
// Strategy: all GEMMs via v_wmma_f32_16x16x32_bf16 with LDS-tiled double-buffered
// pipeline; weights pre-transposed to [N][K] bf16 so B-fragments are contiguous-K
// loads (per ISA 7.12.2 layouts). Attention & policy QK^T computed per (b,head)/(b)
// with WMMA from LDS tiles; softmax via 16-lane shfl_xor reductions matching the
// 16x16 f32 C layout (VGPR r -> rows r/r+8, lane -> column).
#include <hip/hip_runtime.h>
#include <hip/hip_bf16.h>

typedef __attribute__((ext_vector_type(16))) __bf16 bf16x16;
typedef __attribute__((ext_vector_type(8)))  float  f32x8;

union FragBF { bf16x16 v; float4 q[2]; };

#define NEGINF (-1e9f)
enum { ACT_NONE = 0, ACT_MISH = 1 };

__device__ __forceinline__ float softplusf_(float x) {
    return (x > 20.f) ? x : log1pf(__expf(x));
}
__device__ __forceinline__ float mishf(float x) {
    return x * tanhf(softplusf_(x));
}

// ---------------------------------------------------------------------------
// Generic bf16 WMMA GEMM: C[M,N] = act(A[M,K] * Bt[N,K]^T + bias)
// Block tile 128x128, 8 waves (4x2), each wave 32x64 (2x4 accumulators).
// Double-buffered LDS, K-step 32.
// ---------------------------------------------------------------------------
__global__ __launch_bounds__(256) void gemm_bf16_kernel(
    const __bf16* __restrict__ A, const __bf16* __restrict__ Bt,
    const float* __restrict__ bias, float* __restrict__ outF,
    __bf16* __restrict__ outB, int M, int N, int K, int act)
{
    __shared__ __bf16 As[2][128][40];   // [row][k] pad to 40 (80B rows, 16B aligned)
    __shared__ __bf16 Bs[2][128][40];   // [col][k]

    const int tid  = threadIdx.x;
    const int lane = tid & 31;
    const int wid  = tid >> 5;
    const int wm   = (wid >> 1) * 32;   // wave row offset in block tile
    const int wn   = (wid & 1) * 64;    // wave col offset in block tile
    const int m0   = blockIdx.y * 128;
    const int n0   = blockIdx.x * 128;
    const int srow = tid >> 1;          // staging: 2 threads per row
    const int shalf = (tid & 1) * 16;   // 16 bf16 halves
    const int KT = K >> 5;

    const __bf16* gA = A  + (size_t)(m0 + srow) * K + shalf;
    const __bf16* gB = Bt + (size_t)(n0 + srow) * K + shalf;

    // stage k-tile 0
    *(float4*)&As[0][srow][shalf]     = *(const float4*)(gA);
    *(float4*)&As[0][srow][shalf + 8] = *(const float4*)(gA + 8);
    *(float4*)&Bs[0][srow][shalf]     = *(const float4*)(gB);
    *(float4*)&Bs[0][srow][shalf + 8] = *(const float4*)(gB + 8);
    __syncthreads();

    f32x8 acc[2][4];
#pragma unroll
    for (int i = 0; i < 2; ++i)
#pragma unroll
        for (int j = 0; j < 4; ++j)
#pragma unroll
            for (int e = 0; e < 8; ++e) acc[i][j][e] = 0.f;

    // A fragment (16x32 bf16): lane -> row = lane%16; lanes<16: K 0..7 & 16..23,
    // lanes>=16: K 8..15 & 24..31 (two 16B chunks).
    const int arow = wm + (lane & 15);
    const int akb  = (lane < 16) ? 0 : 8;
    // B fragment (32x16): lane -> col = lane%16; lanes<16: K 0..7 & 16..23, lanes>=16: K 8..15 & 24..31.
    const int bcol = wn + (lane & 15);
    const int bkb  = (lane < 16) ? 0 : 8;

    for (int kt = 0; kt < KT; ++kt) {
        const int cur = kt & 1;
        if (kt + 1 < KT) {
            const int nxt = cur ^ 1;
            const __bf16* pa = gA + (size_t)(kt + 1) * 32;
            const __bf16* pb = gB + (size_t)(kt + 1) * 32;
            *(float4*)&As[nxt][srow][shalf]     = *(const float4*)(pa);
            *(float4*)&As[nxt][srow][shalf + 8] = *(const float4*)(pa + 8);
            *(float4*)&Bs[nxt][srow][shalf]     = *(const float4*)(pb);
            *(float4*)&Bs[nxt][srow][shalf + 8] = *(const float4*)(pb + 8);
            if (kt + 2 < KT) {
                __builtin_prefetch(gA + (size_t)(kt + 2) * 32, 0, 1);
                __builtin_prefetch(gB + (size_t)(kt + 2) * 32, 0, 1);
            }
        }
        FragBF a[2];
#pragma unroll
        for (int i = 0; i < 2; ++i) {
            a[i].q[0] = *(const float4*)&As[cur][arow + i * 16][akb];
            a[i].q[1] = *(const float4*)&As[cur][arow + i * 16][akb + 16];
        }
        FragBF b[4];
#pragma unroll
        for (int j = 0; j < 4; ++j) {
            b[j].q[0] = *(const float4*)&Bs[cur][bcol + j * 16][bkb];
            b[j].q[1] = *(const float4*)&Bs[cur][bcol + j * 16][bkb + 16];
        }
#pragma unroll
        for (int i = 0; i < 2; ++i)
#pragma unroll
            for (int j = 0; j < 4; ++j)
                acc[i][j] = __builtin_amdgcn_wmma_f32_16x16x32_bf16(
                    false, a[i].v, false, b[j].v, (short)0, acc[i][j], false, false);
        __syncthreads();
    }

    // Epilogue. C layout: VGPR r: lanes 0-15 -> row r, lanes 16-31 -> row r+8.
    const int hh = (lane < 16) ? 0 : 8;
    const int cl = lane & 15;
#pragma unroll
    for (int i = 0; i < 2; ++i) {
#pragma unroll
        for (int j = 0; j < 4; ++j) {
            const int col = n0 + wn + j * 16 + cl;
            const float bv = bias ? bias[col] : 0.f;
#pragma unroll
            for (int r = 0; r < 8; ++r) {
                const int row = m0 + wm + i * 16 + hh + r;
                float v = acc[i][j][r] + bv;
                if (act == ACT_MISH) v = mishf(v);
                const size_t o = (size_t)row * N + col;
                if (outF) outF[o] = v;
                if (outB) outB[o] = (__bf16)v;
            }
        }
    }
}

// ---------------------------------------------------------------------------
// Attention per (head=blockIdx.x, batch=blockIdx.y): 128 threads (4 waves),
// each wave owns 16 query rows. QK^T (K=32, 1 WMMA step), smolgen bias +
// static bias, row softmax via 16-lane shfl_xor, P (bf16, via LDS) @ V^T.
// ---------------------------------------------------------------------------
__global__ __launch_bounds__(128) void attn_kernel(
    const __bf16* __restrict__ qg, const __bf16* __restrict__ kg,
    const __bf16* __restrict__ vgm, const float* __restrict__ smol,
    const float* __restrict__ stat, __bf16* __restrict__ ogm)
{
    __shared__ __bf16 Qs[64][40];
    __shared__ __bf16 Ks[64][40];
    __shared__ __bf16 Vt[32][72];   // transposed: [d][j]
    __shared__ __bf16 Ps[64][72];   // probabilities

    const int head = blockIdx.x;
    const int b    = blockIdx.y;
    const int tid  = threadIdx.x;
    const int lane = tid & 31;
    const int wid  = tid >> 5;
    const size_t rowbase = ((size_t)b * 64) * 256 + head * 32;

    {   // stage Q,K row-major (row stride 256 in global)
        const int row  = tid >> 1;
        const int half = (tid & 1) * 16;
        const __bf16* pq = qg + rowbase + (size_t)row * 256 + half;
        const __bf16* pk = kg + rowbase + (size_t)row * 256 + half;
        *(float4*)&Qs[row][half]     = *(const float4*)pq;
        *(float4*)&Qs[row][half + 8] = *(const float4*)(pq + 8);
        *(float4*)&Ks[row][half]     = *(const float4*)pk;
        *(float4*)&Ks[row][half + 8] = *(const float4*)(pk + 8);
    }
    for (int e = tid; e < 64 * 32; e += 128) {   // stage V transposed
        const int j = e >> 5, d = e & 31;
        Vt[d][j] = vgm[rowbase + (size_t)j * 256 + d];
    }
    __syncthreads();

    const int r0  = wid * 16;
    const int cl  = lane & 15;
    const int hh  = (lane < 16) ? 0 : 8;
    const int akb = (lane < 16) ? 0 : 8;
    const int bkb = (lane < 16) ? 0 : 8;

    f32x8 lg[4];
#pragma unroll
    for (int j = 0; j < 4; ++j)
#pragma unroll
        for (int e = 0; e < 8; ++e) lg[j][e] = 0.f;

    FragBF aq;
    aq.q[0] = *(const float4*)&Qs[r0 + cl][akb];
    aq.q[1] = *(const float4*)&Qs[r0 + cl][akb + 16];
#pragma unroll
    for (int j = 0; j < 4; ++j) {
        FragBF bk_;
        bk_.q[0] = *(const float4*)&Ks[j * 16 + cl][bkb];
        bk_.q[1] = *(const float4*)&Ks[j * 16 + cl][bkb + 16];
        lg[j] = __builtin_amdgcn_wmma_f32_16x16x32_bf16(
            false, aq.v, false, bk_.v, (short)0, lg[j], false, false);
    }

    const float scale = 0.17677669529663687f;   // 32^-0.5
    const float* vgb = smol + ((size_t)b * 8 + head) * 128;
    float vq[8];
#pragma unroll
    for (int r = 0; r < 8; ++r) vq[r] = vgb[r0 + hh + r];
#pragma unroll
    for (int j = 0; j < 4; ++j) {
        const int col = j * 16 + cl;
        const float vk = vgb[64 + col];
        const float* st = stat + (size_t)head * 4096 + col;
#pragma unroll
        for (int r = 0; r < 8; ++r) {
            const int q = r0 + hh + r;
            lg[j][r] = lg[j][r] * scale + vq[r] * vk + st[(size_t)q * 64];
        }
    }
    // softmax over 64 cols: 4 local vals per lane + 16-lane xor reduction
#pragma unroll
    for (int r = 0; r < 8; ++r) {
        float m = fmaxf(fmaxf(lg[0][r], lg[1][r]), fmaxf(lg[2][r], lg[3][r]));
#pragma unroll
        for (int s = 1; s < 16; s <<= 1) m = fmaxf(m, __shfl_xor(m, s, 32));
        float sum = 0.f;
#pragma unroll
        for (int j = 0; j < 4; ++j) { float e = __expf(lg[j][r] - m); lg[j][r] = e; sum += e; }
#pragma unroll
        for (int s = 1; s < 16; s <<= 1) sum += __shfl_xor(sum, s, 32);
        const float inv = 1.f / sum;
#pragma unroll
        for (int j = 0; j < 4; ++j) lg[j][r] *= inv;
    }
#pragma unroll
    for (int j = 0; j < 4; ++j)
#pragma unroll
        for (int r = 0; r < 8; ++r)
            Ps[r0 + hh + r][j * 16 + cl] = (__bf16)lg[j][r];
    __syncthreads();

    f32x8 oacc[2];
#pragma unroll
    for (int d = 0; d < 2; ++d)
#pragma unroll
        for (int e = 0; e < 8; ++e) oacc[d][e] = 0.f;
#pragma unroll
    for (int kk = 0; kk < 2; ++kk) {
        FragBF pa;
        const int pk0 = kk * 32 + akb;
        pa.q[0] = *(const float4*)&Ps[r0 + cl][pk0];
        pa.q[1] = *(const float4*)&Ps[r0 + cl][pk0 + 16];
#pragma unroll
        for (int dt = 0; dt < 2; ++dt) {
            FragBF vb_;
            const int vk0 = kk * 32 + bkb;
            vb_.q[0] = *(const float4*)&Vt[dt * 16 + cl][vk0];
            vb_.q[1] = *(const float4*)&Vt[dt * 16 + cl][vk0 + 16];
            oacc[dt] = __builtin_amdgcn_wmma_f32_16x16x32_bf16(
                false, pa.v, false, vb_.v, (short)0, oacc[dt], false, false);
        }
    }
#pragma unroll
    for (int dt = 0; dt < 2; ++dt)
#pragma unroll
        for (int r = 0; r < 8; ++r) {
            const int q = r0 + hh + r;
            ogm[rowbase + (size_t)q * 256 + dt * 16 + cl] = (__bf16)oacc[dt][r];
        }
}

// ---------------------------------------------------------------------------
// LayerNorm(h + add): one wave per 256-wide row, 8 rows per 256-thread block.
// Writes f32 master + bf16 copy.
// ---------------------------------------------------------------------------
__global__ __launch_bounds__(256) void ln_kernel(
    const float* hin, const float* addv,
    const float* __restrict__ g, const float* __restrict__ bt,
    float* hout, __bf16* __restrict__ hb)
{
    const int lane = threadIdx.x & 31;
    const int row  = blockIdx.x * 8 + (threadIdx.x >> 5);
    const size_t base = (size_t)row * 256;
    float x[8];
    float s = 0.f;
#pragma unroll
    for (int i = 0; i < 8; ++i) {
        x[i] = hin[base + lane + i * 32] + addv[base + lane + i * 32];
        s += x[i];
    }
#pragma unroll
    for (int m = 1; m < 32; m <<= 1) s += __shfl_xor(s, m, 32);
    const float mean = s * (1.f / 256.f);
    float v = 0.f;
#pragma unroll
    for (int i = 0; i < 8; ++i) { const float d = x[i] - mean; v += d * d; }
#pragma unroll
    for (int m = 1; m < 32; m <<= 1) v += __shfl_xor(v, m, 32);
    const float rstd = rsqrtf(v * (1.f / 256.f) + 1e-5f);
#pragma unroll
    for (int i = 0; i < 8; ++i) {
        const int c = lane + i * 32;
        const float y = (x[i] - mean) * rstd * g[c] + bt[c];
        hout[base + c] = y;
        hb[base + c] = (__bf16)y;
    }
}

// pooled[b][d] = mean_t h[b,t,d]; f32 + bf16 copies.
__global__ __launch_bounds__(256) void pooled_kernel(
    const float* __restrict__ h, float* __restrict__ pf, __bf16* __restrict__ pb)
{
    const int b = blockIdx.x, d = threadIdx.x;
    float s = 0.f;
    for (int t = 0; t < 64; ++t) s += h[((size_t)b * 64 + t) * 256 + d];
    s *= (1.f / 64.f);
    pf[(size_t)b * 256 + d] = s;
    pb[(size_t)b * 256 + d] = (__bf16)s;
}

// (B,C,8,8) -> tokens [B*64][128] bf16, K padded 112 -> 128 with zeros.
__global__ __launch_bounds__(256) void tokenize_kernel(
    const float* __restrict__ x, __bf16* __restrict__ xtok)
{
    const int idx = blockIdx.x * 256 + threadIdx.x;
    const int m = idx >> 7, c = idx & 127;
    const int b = m >> 6, t = m & 63;
    const float v = (c < 112) ? x[((size_t)b * 112 + c) * 64 + t] : 0.f;
    xtok[idx] = (__bf16)v;
}

// W[K][N] f32 -> Wt[Npad][Kpad] bf16 (transposed, zero-padded).
__global__ __launch_bounds__(256) void prep_wt_kernel(
    const float* __restrict__ W, __bf16* __restrict__ Wt, int K, int N, int Kp, int Np)
{
    const int idx = blockIdx.x * 256 + threadIdx.x;
    if (idx >= Np * Kp) return;
    const int n = idx / Kp, k = idx - n * Kp;
    Wt[idx] = (n < N && k < K) ? (__bf16)W[(size_t)k * N + n] : (__bf16)0.f;
}

__global__ void pad_bias_kernel(const float* __restrict__ b, float* __restrict__ o, int n, int npad)
{
    const int i = threadIdx.x;
    if (i < npad) o[i] = (i < n) ? b[i] : 0.f;
}

// ---------------------------------------------------------------------------
// Policy head per batch: lft = pq @ pk^T / sqrt(128) via WMMA (K=128, 4 steps),
// then on-the-fly LC0 move-table gather + promo plane masking.
// ---------------------------------------------------------------------------
__global__ __launch_bounds__(128) void policy_kernel(
    const __bf16* __restrict__ pqg, const __bf16* __restrict__ pkg,
    const float* __restrict__ upo, float* __restrict__ out)
{
    __shared__ __bf16 PQ[64][136];
    __shared__ __bf16 PK[64][136];
    __shared__ float LFT[64][68];
    const int b = blockIdx.x;
    const int tid = threadIdx.x;
    const int lane = tid & 31;
    const int wid = tid >> 5;
    {
        const int row = tid >> 1;
        const int half = (tid & 1) * 64;
        const __bf16* pq = pqg + ((size_t)(b * 64 + row)) * 128 + half;
        const __bf16* pk = pkg + ((size_t)(b * 64 + row)) * 128 + half;
#pragma unroll
        for (int c = 0; c < 8; ++c) {
            *(float4*)&PQ[row][half + c * 8] = *(const float4*)(pq + c * 8);
            *(float4*)&PK[row][half + c * 8] = *(const float4*)(pk + c * 8);
        }
    }
    __syncthreads();
    const int r0 = wid * 16;
    const int cl = lane & 15;
    const int hh = (lane < 16) ? 0 : 8;
    const int akb = (lane < 16) ? 0 : 8;
    const int bkb = (lane < 16) ? 0 : 8;
    f32x8 acc[4];
#pragma unroll
    for (int j = 0; j < 4; ++j)
#pragma unroll
        for (int e = 0; e < 8; ++e) acc[j][e] = 0.f;
    for (int kt = 0; kt < 4; ++kt) {
        FragBF a;
        a.q[0] = *(const float4*)&PQ[r0 + cl][kt * 32 + akb];
        a.q[1] = *(const float4*)&PQ[r0 + cl][kt * 32 + akb + 16];
#pragma unroll
        for (int j = 0; j < 4; ++j) {
            FragBF bb;
            bb.q[0] = *(const float4*)&PK[j * 16 + cl][kt * 32 + bkb];
            bb.q[1] = *(const float4*)&PK[j * 16 + cl][kt * 32 + bkb + 16];
            acc[j] = __builtin_amdgcn_wmma_f32_16x16x32_bf16(
                false, a.v, false, bb.v, (short)0, acc[j], false, false);
        }
    }
    const float scale = 0.08838834764831845f;   // 128^-0.5
#pragma unroll
    for (int j = 0; j < 4; ++j)
#pragma unroll
        for (int r = 0; r < 8; ++r)
            LFT[r0 + hh + r][j * 16 + cl] = acc[j][r] * scale;
    __syncthreads();

    const int dr8[8] = {1, 1, 0, -1, -1, -1, 0, 1};
    const int dc8[8] = {0, 1, 1, 1, 0, -1, -1, -1};
    const int kr8[8] = {2, 1, -1, -2, -2, -1, 1, 2};
    const int kc8[8] = {1, 2, 2, 1, -1, -2, -2, -1};
    for (int idx = tid; idx < 64 * 73; idx += 128) {
        const int q = idx / 73;
        const int p = idx - q * 73;
        const int r = q >> 3, c = q & 7;
        float val;
        if (p < 64) {
            int rr, cc;
            if (p < 56) {
                const int dir = p / 7, dist = p - dir * 7 + 1;
                rr = r + dr8[dir] * dist; cc = c + dc8[dir] * dist;
            } else {
                rr = r + kr8[p - 56]; cc = c + kc8[p - 56];
            }
            const bool ok = (rr >= 0 && rr < 8 && cc >= 0 && cc < 8);
            val = ok ? LFT[q][rr * 8 + cc] : NEGINF;
        } else {
            val = (r == 6) ? upo[((size_t)(b * 64 + q)) * 128 + (p - 64)] : NEGINF;
        }
        out[(size_t)b * 4672 + idx] = val;
    }
}

// Scalar heads (value/vol/ml) per batch row; tiny, pure VALU.
__global__ __launch_bounds__(128) void heads_kernel(
    const float* __restrict__ pooled,
    const float* __restrict__ v1w, const float* __restrict__ v1b,
    const float* __restrict__ v2w, const float* __restrict__ v2b,
    const float* __restrict__ vo1w, const float* __restrict__ vo1b,
    const float* __restrict__ vo2w, const float* __restrict__ vo2b,
    const float* __restrict__ m1w, const float* __restrict__ m1b,
    const float* __restrict__ m2w, const float* __restrict__ m2b,
    float* __restrict__ out)
{
    constexpr size_t V_OFF = (size_t)512 * 4672;
    constexpr size_t VOL_OFF = V_OFF + 512 * 3;
    constexpr size_t ML_OFF = VOL_OFF + 512 * 3;
    __shared__ float ps[256], t1[128], t2[64], t3[32];
    const int b = blockIdx.x, tid = threadIdx.x;
    ps[tid] = pooled[(size_t)b * 256 + tid];
    ps[tid + 128] = pooled[(size_t)b * 256 + tid + 128];
    __syncthreads();
    {
        float s = v1b[tid];
        for (int k = 0; k < 256; ++k) s += ps[k] * v1w[(size_t)k * 128 + tid];
        t1[tid] = mishf(s);
    }
    if (tid < 64) {
        float s = vo1b[tid];
        for (int k = 0; k < 256; ++k) s += ps[k] * vo1w[(size_t)k * 64 + tid];
        t2[tid] = mishf(s);
    }
    if (tid < 32) {
        float s = m1b[tid];
        for (int k = 0; k < 256; ++k) s += ps[k] * m1w[(size_t)k * 32 + tid];
        t3[tid] = mishf(s);
    }
    __syncthreads();
    if (tid < 3) {
        float s = v2b[tid];
        for (int k = 0; k < 128; ++k) s += t1[k] * v2w[k * 3 + tid];
        out[V_OFF + (size_t)b * 3 + tid] = s;
    } else if (tid < 6) {
        const int c = tid - 3;
        float s = vo2b[c];
        for (int k = 0; k < 64; ++k) s += t2[k] * vo2w[k * 3 + c];
        out[VOL_OFF + (size_t)b * 3 + c] = softplusf_(s);
    } else if (tid == 6) {
        float s = m2b[0];
        for (int k = 0; k < 32; ++k) s += t3[k] * m2w[k];
        out[ML_OFF + b] = s;
    }
}

// ---------------------------------------------------------------------------
extern "C" void kernel_launch(void* const* d_in, const int* in_sizes, int n_in,
                              void* d_out, int out_size, void* d_ws, size_t ws_size,
                              hipStream_t stream)
{
    (void)in_sizes; (void)n_in; (void)out_size; (void)ws_size;
    const float* x       = (const float*)d_in[0];
    const float* embed_w = (const float*)d_in[1];
    const float* embed_b = (const float*)d_in[2];
    const float* wq = (const float*)d_in[3];  const float* bq = (const float*)d_in[4];
    const float* wk = (const float*)d_in[5];  const float* bk = (const float*)d_in[6];
    const float* wv = (const float*)d_in[7];  const float* bv = (const float*)d_in[8];
    const float* wo = (const float*)d_in[9];  const float* bo = (const float*)d_in[10];
    const float* smol_static = (const float*)d_in[11];
    const float* cw  = (const float*)d_in[12]; const float* cb  = (const float*)d_in[13];
    const float* g1w = (const float*)d_in[14]; const float* g1b = (const float*)d_in[15];
    const float* g2w = (const float*)d_in[16]; const float* g2b = (const float*)d_in[17];
    const float* ln1g = (const float*)d_in[18]; const float* ln1b = (const float*)d_in[19];
    const float* ln2g = (const float*)d_in[20]; const float* ln2b = (const float*)d_in[21];
    const float* f1w = (const float*)d_in[22]; const float* f1b = (const float*)d_in[23];
    const float* f2w = (const float*)d_in[24]; const float* f2b = (const float*)d_in[25];
    const float* pqw = (const float*)d_in[26]; const float* pqb = (const float*)d_in[27];
    const float* pkw = (const float*)d_in[28]; const float* pkb = (const float*)d_in[29];
    const float* upw = (const float*)d_in[30]; const float* upb = (const float*)d_in[31];
    const float* v1w = (const float*)d_in[32]; const float* v1b = (const float*)d_in[33];
    const float* v2w = (const float*)d_in[34]; const float* v2b = (const float*)d_in[35];
    const float* vo1w = (const float*)d_in[36]; const float* vo1b = (const float*)d_in[37];
    const float* vo2w = (const float*)d_in[38]; const float* vo2b = (const float*)d_in[39];
    const float* m1w = (const float*)d_in[40]; const float* m1b = (const float*)d_in[41];
    const float* m2w = (const float*)d_in[42]; const float* m2b = (const float*)d_in[43];
    float* out = (float*)d_out;

    size_t off = 0;
    auto alloc = [&](size_t bytes) -> void* {
        void* p = (char*)d_ws + off;
        off += (bytes + 255) & ~(size_t)255;
        return p;
    };
    const int M = 32768;   // B*T rows
    __bf16* xtok = (__bf16*)alloc((size_t)M * 128 * 2);
    __bf16* hb   = (__bf16*)alloc((size_t)M * 256 * 2);
    float*  hF   = (float*)alloc((size_t)M * 256 * 4);
    __bf16* qB   = (__bf16*)alloc((size_t)M * 256 * 2);
    __bf16* kB   = (__bf16*)alloc((size_t)M * 256 * 2);
    __bf16* vB   = (__bf16*)alloc((size_t)M * 256 * 2);
    __bf16* oB   = (__bf16*)alloc((size_t)M * 256 * 2);
    float*  og   = (float*)alloc((size_t)M * 256 * 4);
    __bf16* ffB  = (__bf16*)alloc((size_t)M * 512 * 2);
    float*  poolF = (float*)alloc(512 * 256 * 4);
    __bf16* poolB = (__bf16*)alloc(512 * 256 * 2);
    __bf16* hcB  = (__bf16*)alloc(512 * 256 * 2);
    __bf16* g1B  = (__bf16*)alloc(512 * 256 * 2);
    float*  vgF  = (float*)alloc(512 * 1024 * 4);
    float*  upo  = (float*)alloc((size_t)M * 128 * 4);
    float*  upbPad = (float*)alloc(128 * 4);
    __bf16* wtEmbed = (__bf16*)alloc(256 * 128 * 2);
    __bf16* wtQ  = (__bf16*)alloc((size_t)6 * 65536 * 2);
    __bf16* wtK  = (__bf16*)alloc((size_t)6 * 65536 * 2);
    __bf16* wtV  = (__bf16*)alloc((size_t)6 * 65536 * 2);
    __bf16* wtO  = (__bf16*)alloc((size_t)6 * 65536 * 2);
    __bf16* wtCW = (__bf16*)alloc((size_t)6 * 65536 * 2);
    __bf16* wtG1 = (__bf16*)alloc((size_t)6 * 65536 * 2);
    __bf16* wtG2 = (__bf16*)alloc((size_t)6 * 262144 * 2);
    __bf16* wtF1 = (__bf16*)alloc((size_t)6 * 131072 * 2);
    __bf16* wtF2 = (__bf16*)alloc((size_t)6 * 131072 * 2);
    __bf16* wtPQ = (__bf16*)alloc(128 * 256 * 2);
    __bf16* wtPK = (__bf16*)alloc(128 * 256 * 2);
    __bf16* wtUP = (__bf16*)alloc(128 * 256 * 2);

    auto prep = [&](const float* W, __bf16* Wt, int K, int N, int Kp, int Np) {
        const int total = Np * Kp;
        prep_wt_kernel<<<dim3((total + 255) / 256), dim3(256), 0, stream>>>(W, Wt, K, N, Kp, Np);
    };
    prep(embed_w, wtEmbed, 112, 256, 128, 256);
    for (int l = 0; l < 6; ++l) {
        prep(wq + (size_t)l * 65536, wtQ + (size_t)l * 65536, 256, 256, 256, 256);
        prep(wk + (size_t)l * 65536, wtK + (size_t)l * 65536, 256, 256, 256, 256);
        prep(wv + (size_t)l * 65536, wtV + (size_t)l * 65536, 256, 256, 256, 256);
        prep(wo + (size_t)l * 65536, wtO + (size_t)l * 65536, 256, 256, 256, 256);
        prep(cw + (size_t)l * 65536, wtCW + (size_t)l * 65536, 256, 256, 256, 256);
        prep(g1w + (size_t)l * 65536, wtG1 + (size_t)l * 65536, 256, 256, 256, 256);
        prep(g2w + (size_t)l * 262144, wtG2 + (size_t)l * 262144, 256, 1024, 256, 1024);
        prep(f1w + (size_t)l * 131072, wtF1 + (size_t)l * 131072, 256, 512, 256, 512);
        prep(f2w + (size_t)l * 131072, wtF2 + (size_t)l * 131072, 512, 256, 512, 256);
    }
    prep(pqw, wtPQ, 256, 128, 256, 128);
    prep(pkw, wtPK, 256, 128, 256, 128);
    prep(upw, wtUP, 256, 9, 256, 128);
    pad_bias_kernel<<<dim3(1), dim3(128), 0, stream>>>(upb, upbPad, 9, 128);
    tokenize_kernel<<<dim3(M * 128 / 256), dim3(256), 0, stream>>>(x, xtok);

    auto gemm = [&](const __bf16* A, const __bf16* Bt, const float* bias,
                    float* oF, __bf16* oB2, int Mm, int Nn, int Kk, int act) {
        gemm_bf16_kernel<<<dim3(Nn / 128, Mm / 128), dim3(256), 0, stream>>>(
            A, Bt, bias, oF, oB2, Mm, Nn, Kk, act);
    };

    gemm(xtok, wtEmbed, embed_b, hF, hb, M, 256, 128, ACT_NONE);

    for (int l = 0; l < 6; ++l) {
        pooled_kernel<<<dim3(512), dim3(256), 0, stream>>>(hF, poolF, poolB);
        gemm(poolB, wtCW + (size_t)l * 65536, cb + l * 256, nullptr, hcB, 512, 256, 256, ACT_NONE);
        gemm(hcB, wtG1 + (size_t)l * 65536, g1b + l * 256, nullptr, g1B, 512, 256, 256, ACT_MISH);
        gemm(g1B, wtG2 + (size_t)l * 262144, g2b + l * 1024, vgF, nullptr, 512, 1024, 256, ACT_NONE);
        gemm(hb, wtQ + (size_t)l * 65536, bq + l * 256, nullptr, qB, M, 256, 256, ACT_NONE);
        gemm(hb, wtK + (size_t)l * 65536, bk + l * 256, nullptr, kB, M, 256, 256, ACT_NONE);
        gemm(hb, wtV + (size_t)l * 65536, bv + l * 256, nullptr, vB, M, 256, 256, ACT_NONE);
        attn_kernel<<<dim3(8, 512), dim3(128), 0, stream>>>(
            qB, kB, vB, vgF, smol_static + (size_t)l * 32768, oB);
        gemm(oB, wtO + (size_t)l * 65536, bo + l * 256, og, nullptr, M, 256, 256, ACT_NONE);
        ln_kernel<<<dim3(M / 8), dim3(256), 0, stream>>>(hF, og, ln1g + l * 256, ln1b + l * 256, hF, hb);
        gemm(hb, wtF1 + (size_t)l * 131072, f1b + l * 512, nullptr, ffB, M, 512, 256, ACT_MISH);
        gemm(ffB, wtF2 + (size_t)l * 131072, f2b + l * 256, og, nullptr, M, 256, 512, ACT_NONE);
        ln_kernel<<<dim3(M / 8), dim3(256), 0, stream>>>(hF, og, ln2g + l * 256, ln2b + l * 256, hF, hb);
    }

    // heads
    gemm(hb, wtPQ, pqb, nullptr, qB, M, 128, 256, ACT_NONE);
    gemm(hb, wtPK, pkb, nullptr, kB, M, 128, 256, ACT_NONE);
    gemm(hb, wtUP, upbPad, upo, nullptr, M, 128, 256, ACT_NONE);
    policy_kernel<<<dim3(512), dim3(128), 0, stream>>>(qB, kB, upo, out);
    pooled_kernel<<<dim3(512), dim3(256), 0, stream>>>(hF, poolF, poolB);
    heads_kernel<<<dim3(512), dim3(128), 0, stream>>>(
        poolF, v1w, v1b, v2w, v2b, vo1w, vo1b, vo2w, vo2b, m1w, m1b, m2w, m2b, out);
}
